// MultiEDLSTM_25082609009017
// MI455X (gfx1250) — compile-verified, matrix operand order checked by reading, not run on hardware
//
#include <hip/hip_runtime.h>
#include <hip/hip_bf16.h>

// ---------------------------------------------------------------------------
// MultiEDLSTM on gfx1250: WMMA bf16 (f32 accum) for all GEMMs, persistent
// grid-synced kernels for the sequential LSTM recurrences, LDS-resident
// recurrent weights, and TDM (tensor_load_to_lds) double-buffered prefetch
// of the decoder context tiles.
// Shapes: B=32, S=1024, WIN=1024, E=128, H=512, L=3
// ---------------------------------------------------------------------------

#define B_ 32
#define S_ 1024
#define WIN_ 1024
#define E_ 128
#define H_ 512
#define NBLK_REC 32   // persistent blocks for recurrences (one 16-wide H slice each)

typedef __attribute__((ext_vector_type(16))) __bf16 v16bf;
typedef __attribute__((ext_vector_type(8)))  float  v8f;
typedef __attribute__((ext_vector_type(4)))  unsigned v4u;
typedef __attribute__((ext_vector_type(8)))  unsigned v8u;

__device__ __forceinline__ v8f wmma_bf16(v16bf a, v16bf b, v8f c) {
  return __builtin_amdgcn_wmma_f32_16x16x32_bf16(false, a, false, b, (short)0, c,
                                                 false, false);
}

// A-fragment (16x32 bf16, MxK), ISA 7.12.2 layout:
// lane L: row = L&15, half = L>>4; VGPR0..3 hold K = 8*half + {0..7},
// VGPR4..7 hold K = 16 + 8*half + {0..7}.  Source row-major [row][K], ld=K-stride.
// Also used for B tiles (32x16 KxN) from [n][k]-major weights with lane -> N.
__device__ __forceinline__ v16bf frag_ld_bf16(const __bf16* base, int ld) {
  const int lane = threadIdx.x & 31;
  const __bf16* p = base + (lane & 15) * ld + ((lane >> 4) << 3);
  v16bf r;
#pragma unroll
  for (int i = 0; i < 8; ++i) { r[i] = p[i]; r[i + 8] = p[i + 16]; }
  return r;
}

// Same pattern from f32 storage with in-register conversion (one-shot GEMMs only).
__device__ __forceinline__ v16bf frag_ld_f32(const float* base, int ld) {
  const int lane = threadIdx.x & 31;
  const float* p = base + (lane & 15) * ld + ((lane >> 4) << 3);
  v16bf r;
#pragma unroll
  for (int i = 0; i < 8; ++i) { r[i] = (__bf16)p[i]; r[i + 8] = (__bf16)p[i + 16]; }
  return r;
}

__device__ __forceinline__ float sigm(float x) { return 1.0f / (1.0f + __expf(-x)); }

// ---------------------------------------------------------------------------
// Tensor Data Mover: DMA a 2D tile (tile_y rows x tile_x bf16 elements, rows
// row_stride_elems apart) from global memory into contiguous LDS at lds_off.
// D# built per ISA §8 (groups 0/1; groups 2/3 zero quads -> dims >=2 unused).
// Issue from ONE wave; completion via s_wait_tensorcnt.  EXEC is ignored.
// ---------------------------------------------------------------------------
__device__ __forceinline__ void tdm_stage_2d(unsigned lds_off, const void* gptr,
                                             unsigned tile_x, unsigned tile_y,
                                             unsigned row_stride_elems) {
  const unsigned long long ga = (unsigned long long)gptr;
  v4u g0;
  g0[0] = 1u;                                   // count=1, user descriptor
  g0[1] = lds_off;                              // LDS byte address
  g0[2] = (unsigned)ga;                         // global addr [31:0]
  g0[3] = (unsigned)((ga >> 32) & 0x1FFFFFFu)   // global addr [56:32]
          | (2u << 30);                         // type = 2 ("image")
  v8u g1;
  g1[0] = (1u << 16);                           // data_size=1 (2 bytes), mask=0
  g1[1] = (tile_x & 0xFFFFu) << 16;             // tensor_dim0[15:0]
  g1[2] = ((tile_x >> 16) & 0xFFFFu)            // tensor_dim0[31:16]
          | ((tile_y & 0xFFFFu) << 16);         // tensor_dim1[15:0]
  g1[3] = ((tile_y >> 16) & 0xFFFFu)            // tensor_dim1[31:16]
          | ((tile_x & 0xFFFFu) << 16);         // tile_dim0
  g1[4] = (tile_y & 0xFFFFu);                   // tile_dim1 (tile_dim2=0)
  g1[5] = row_stride_elems;                     // tensor_dim0_stride[31:0]
  g1[6] = 0u;                                   // stride hi / dim1_stride lo
  g1[7] = 0u;
  const v4u gz = {0u, 0u, 0u, 0u};
  asm volatile("tensor_load_to_lds %0, %1, %2, %3"
               :: "s"(g0), "s"(g1), "s"(gz), "s"(gz)
               : "memory");
}

// Raw LDS byte offset of a __shared__ object: low 32 bits of the generic
// address (LDS aperture base occupies the high word on gfx9+).
__device__ __forceinline__ unsigned lds_offset_of(const void* p) {
  return (unsigned)(unsigned long long)p;
}

// ---------------------------------------------------------------------------
// One-shot f32 -> bf16 weight conversion / repack kernels.
// ---------------------------------------------------------------------------
__global__ void cvt_kernel(const float* __restrict__ src, __bf16* __restrict__ dst,
                           int n) {
  const int i = blockIdx.x * 256 + threadIdx.x;
  if (i < n) dst[i] = (__bf16)src[i];
}

// Wih_d is (4H, 1+H); repack columns 1..H into a tight, aligned [4H][H] bf16.
__global__ void repack_wihd_kernel(const float* __restrict__ src,
                                   __bf16* __restrict__ dst) {
  const int i = blockIdx.x * 256 + threadIdx.x;  // 4H*H
  const int n = i >> 9, k = i & (H_ - 1);
  dst[i] = (__bf16)src[(size_t)n * (1 + H_) + 1 + k];
}

// ---------------------------------------------------------------------------
// Generic row-tile GEMM:  O = relu(A @ W^T + bias), O stored bf16.
// grid.x = M/16 tiles; 8 waves each own one 16-wide N tile (N == 128).
// ---------------------------------------------------------------------------
template <bool A_BF16>
__global__ __launch_bounds__(256) void gemm_relu_kernel(
    const void* __restrict__ A, int lda, const __bf16* __restrict__ W, int ldb,
    const float* __restrict__ bias, __bf16* __restrict__ O, int ldo, int K) {
  const int mt = blockIdx.x;
  const int wave = threadIdx.x >> 5, lane = threadIdx.x & 31;
  const int lh = lane >> 4, nl = lane & 15;
  const int n0 = wave * 16;
  const float bv = bias[n0 + nl];
  v8f acc;
#pragma unroll
  for (int v = 0; v < 8; ++v) acc[v] = bv;
  const int nk = K >> 5;
  for (int kc = 0; kc < nk; ++kc) {
    v16bf a;
    if constexpr (A_BF16)
      a = frag_ld_bf16((const __bf16*)A + (size_t)mt * 16 * lda + kc * 32, lda);
    else
      a = frag_ld_f32((const float*)A + (size_t)mt * 16 * lda + kc * 32, lda);
    v16bf b = frag_ld_bf16(W + (size_t)n0 * ldb + kc * 32, ldb);
    acc = wmma_bf16(a, b, acc);
  }
#pragma unroll
  for (int v = 0; v < 8; ++v) {
    const int m = mt * 16 + lh * 8 + v;
    float x = acc[v];
    x = x > 0.f ? x : 0.f;
    O[(size_t)m * ldo + n0 + nl] = (__bf16)x;
  }
}

// ---------------------------------------------------------------------------
// Attention logits: a[r,l] = a1[r,:] . W_a2[l,:] + b_a2[l]   (L=3)
// ---------------------------------------------------------------------------
__global__ __launch_bounds__(256) void attn_logits_kernel(
    const __bf16* __restrict__ a1, const float* __restrict__ W_a2,
    const float* __restrict__ b_a2, float* __restrict__ a) {
  const int r = blockIdx.x * 256 + threadIdx.x;
  if (r >= B_ * S_) return;
  float s0 = b_a2[0], s1 = b_a2[1], s2 = b_a2[2];
  const __bf16* row = a1 + (size_t)r * E_;
#pragma unroll 8
  for (int e = 0; e < E_; ++e) {
    const float x = (float)row[e];
    s0 += x * W_a2[e];
    s1 += x * W_a2[E_ + e];
    s2 += x * W_a2[2 * E_ + e];
  }
  a[(size_t)r * 3 + 0] = s0;
  a[(size_t)r * 3 + 1] = s1;
  a[(size_t)r * 3 + 2] = s2;
}

// ---------------------------------------------------------------------------
// Softmax over TIME (axis=1), in place.  One block per (b, l) pair (96 total).
// ---------------------------------------------------------------------------
__global__ __launch_bounds__(256) void softmax_time_kernel(float* __restrict__ a) {
  const int b = blockIdx.x / 3, l = blockIdx.x % 3;
  float* base = a + (size_t)b * S_ * 3 + l;
  __shared__ float red[256];
  const int tid = threadIdx.x;
  float mx = -1e30f;
  for (int t = tid; t < S_; t += 256) mx = fmaxf(mx, base[(size_t)t * 3]);
  red[tid] = mx;
  __syncthreads();
  for (int s = 128; s > 0; s >>= 1) {
    if (tid < s) red[tid] = fmaxf(red[tid], red[tid + s]);
    __syncthreads();
  }
  mx = red[0];
  __syncthreads();
  float sum = 0.f;
  for (int t = tid; t < S_; t += 256) sum += __expf(base[(size_t)t * 3] - mx);
  red[tid] = sum;
  __syncthreads();
  for (int s = 128; s > 0; s >>= 1) {
    if (tid < s) red[tid] += red[tid + s];
    __syncthreads();
  }
  const float inv = 1.0f / red[0];
  for (int t = tid; t < S_; t += 256)
    base[(size_t)t * 3] = __expf(base[(size_t)t * 3] - mx) * inv;
}

// ---------------------------------------------------------------------------
// Init: zero sync counters, p ping-pong, broadcast learned h0 states (bf16).
// ---------------------------------------------------------------------------
__global__ void init_kernel(const float* __restrict__ enc_h0,
                            const float* __restrict__ dec_h0,
                            __bf16* __restrict__ he0, __bf16* __restrict__ hd0,
                            float* __restrict__ p0, float* __restrict__ p1,
                            unsigned* __restrict__ cnts) {
  const int idx = blockIdx.x * 256 + threadIdx.x;
  if (idx < B_ * H_) {
    he0[idx] = (__bf16)enc_h0[idx & (H_ - 1)];
    hd0[idx] = (__bf16)dec_h0[idx & (H_ - 1)];
  }
  if (idx < B_) { p0[idx] = 0.f; p1[idx] = 0.f; }
  if (idx < 16) cnts[idx] = 0u;
}

// ---------------------------------------------------------------------------
// Encoder LSTM: persistent, NBLK_REC blocks, block owns h columns
// [blk*16, blk*16+16).  Weight slices staged in LDS once.
// ---------------------------------------------------------------------------
__global__ __launch_bounds__(256) void enc_lstm_kernel(
    const __bf16* __restrict__ embed, const __bf16* __restrict__ Wih,
    const __bf16* __restrict__ Whh, const float* __restrict__ bih,
    const float* __restrict__ bhh, const float* __restrict__ c0,
    const int* __restrict__ lengths, __bf16* __restrict__ h0buf,
    __bf16* __restrict__ h1buf, float* __restrict__ enc_out,
    unsigned* __restrict__ cnt) {
  __shared__ __bf16 whh_sl[64 * H_];     // 64 KB: this block's Whh slice
  __shared__ __bf16 wih_sl[64 * E_];     // 16 KB: this block's Wih slice
  __shared__ float gates_sl[4][16][33];  // [gate][ncol][batch]
  __shared__ float c_sl[16][32];         // cell state, private to this block
  const int blk = blockIdx.x;
  const int tid = threadIdx.x, wave = tid >> 5, lane = tid & 31;
  const int mt = wave & 1, g = wave >> 1;  // M tile / gate (i,f,g,o)
  const int lh = lane >> 4, nl = lane & 15;
  const int ncol0 = blk * 16;
  const int n_gate = g * H_ + ncol0 + nl;
  const float bias_n = bih[n_gate] + bhh[n_gate];
  // stage weight slices: local row lr -> global gate row (lr>>4)*H + ncol0 + (lr&15)
  {
    const int lr = tid >> 2, q = tid & 3;
    const int gr = (lr >> 4) * H_ + ncol0 + (lr & 15);
    const __bf16* s1 = Whh + (size_t)gr * H_ + q * 128;
    __bf16* d1 = whh_sl + lr * H_ + q * 128;
#pragma unroll
    for (int i = 0; i < 128; ++i) d1[i] = s1[i];
    const __bf16* s2 = Wih + (size_t)gr * E_ + q * 32;
    __bf16* d2 = wih_sl + lr * E_ + q * 32;
#pragma unroll
    for (int i = 0; i < 32; ++i) d2[i] = s2[i];
  }
  for (int i = tid; i < 512; i += 256) c_sl[i >> 5][i & 31] = c0[ncol0 + (i >> 5)];
  __syncthreads();
  const int SE = S_ * E_;
  for (int t = 0; t < S_; ++t) {
    const __bf16* hc = (t & 1) ? h1buf : h0buf;
    __bf16* hn = (t & 1) ? h0buf : h1buf;
    v8f acc;
#pragma unroll
    for (int v = 0; v < 8; ++v) acc[v] = bias_n;
    const __bf16* abase_e = embed + (size_t)(mt * 16) * SE + t * E_;
    __builtin_prefetch(abase_e + E_, 0, 1);  // next timestep embed row
#pragma unroll
    for (int kc = 0; kc < E_ / 32; ++kc)
      acc = wmma_bf16(frag_ld_bf16(abase_e + kc * 32, SE),
                      frag_ld_bf16(wih_sl + (g * 16) * E_ + kc * 32, E_), acc);
    const __bf16* abase_h = hc + mt * 16 * H_;
#pragma unroll 4
    for (int kc = 0; kc < H_ / 32; ++kc)
      acc = wmma_bf16(frag_ld_bf16(abase_h + kc * 32, H_),
                      frag_ld_bf16(whh_sl + (g * 16) * H_ + kc * 32, H_), acc);
#pragma unroll
    for (int v = 0; v < 8; ++v) gates_sl[g][nl][mt * 16 + lh * 8 + v] = acc[v];
    __syncthreads();
#pragma unroll
    for (int it = 0; it < 2; ++it) {
      const int idx = tid + it * 256;
      const int m = idx & 31, nc = idx >> 5;
      const float iv = gates_sl[0][nc][m], fv = gates_sl[1][nc][m];
      const float gv = gates_sl[2][nc][m], ov = gates_sl[3][nc][m];
      float c = sigm(fv) * c_sl[nc][m] + sigm(iv) * tanhf(gv);
      const float h = sigm(ov) * tanhf(c);
      c_sl[nc][m] = c;
      hn[m * H_ + ncol0 + nc] = (__bf16)h;
      enc_out[(size_t)m * (S_ * H_) + (size_t)t * H_ + ncol0 + nc] =
          (t < lengths[m]) ? h : 0.f;
    }
    __threadfence();
    __syncthreads();
    if (tid == 0) {
      __hip_atomic_fetch_add(cnt, 1u, __ATOMIC_RELAXED, __HIP_MEMORY_SCOPE_AGENT);
      const unsigned tgt = (unsigned)(t + 1) * (unsigned)NBLK_REC;
      while (__hip_atomic_load(cnt, __ATOMIC_RELAXED, __HIP_MEMORY_SCOPE_AGENT) < tgt)
        __builtin_amdgcn_s_sleep(2);
    }
    __syncthreads();
  }
}

// ---------------------------------------------------------------------------
// context[b,t,:] = sum_{i<3} attn[b,t,i] * enc_out[b,t-i,:]  -> bf16
// ---------------------------------------------------------------------------
__global__ __launch_bounds__(256) void ctx_kernel(const float* __restrict__ attn,
                                                  const float* __restrict__ enc_out,
                                                  __bf16* __restrict__ ctx) {
  const size_t idx = (size_t)blockIdx.x * 256 + threadIdx.x;  // B*S*H
  const size_t bt = idx >> 9;
  const int t = (int)(bt & (S_ - 1));
  const float* at = attn + bt * 3;
  const float* eo = enc_out + idx;
  float s = at[0] * eo[0];
  if (t >= 1) s += at[1] * eo[-H_];
  if (t >= 2) s += at[2] * eo[-2 * H_];
  ctx[idx] = (__bf16)s;
}

// ---------------------------------------------------------------------------
// Decoder LSTM: persistent.  gates = p*wcol0 + ctx_t@Wih[:,1:]^T + h@Whh^T + b.
// Whh slice LDS-resident; ctx_t tiles DMA'd by the Tensor Data Mover into a
// double-buffered LDS staging area, prefetched one step ahead (TDM for t+1 is
// issued before consuming t, overlapping the DMA with the whole step).
// Block 0 additionally computes p = relu(h@W_o1^T + b_o1)@W_o2^T + b_o2.
// ---------------------------------------------------------------------------
__global__ __launch_bounds__(256) void dec_lstm_kernel(
    const __bf16* __restrict__ ctx, const float* __restrict__ Wih_full,
    const __bf16* __restrict__ Wih_ctx, const __bf16* __restrict__ Whh,
    const float* __restrict__ bih, const float* __restrict__ bhh,
    const float* __restrict__ c0, const __bf16* __restrict__ W_o1,
    const float* __restrict__ b_o1, const float* __restrict__ W_o2,
    const float* __restrict__ b_o2, const float* __restrict__ mask,
    __bf16* __restrict__ h0buf, __bf16* __restrict__ h1buf,
    float* __restrict__ p0buf, float* __restrict__ p1buf,
    float* __restrict__ out, unsigned* __restrict__ cntA,
    unsigned* __restrict__ cntB) {
  __shared__ __bf16 whh_sl[64 * H_];        // 64 KB: this block's Whh slice
  __shared__ __bf16 ctx_sl[2][B_ * H_];     // 64 KB: TDM-staged ctx_t, 2 buffers
  __shared__ float gates_sl[4][16][33];
  __shared__ float c_sl[16][32];
  __shared__ float o1_sl[32][129];  // block 0 only
  const int blk = blockIdx.x;
  const int tid = threadIdx.x, wave = tid >> 5, lane = tid & 31;
  const int mt = wave & 1, g = wave >> 1;
  const int lh = lane >> 4, nl = lane & 15;
  const int ncol0 = blk * 16;
  const int n_gate = g * H_ + ncol0 + nl;
  const float bias_n = bih[n_gate] + bhh[n_gate];
  const float wp0 = Wih_full[(size_t)n_gate * (1 + H_)];  // weight on scalar p
  {
    const int lr = tid >> 2, q = tid & 3;
    const int gr = (lr >> 4) * H_ + ncol0 + (lr & 15);
    const __bf16* s1 = Whh + (size_t)gr * H_ + q * 128;
    __bf16* d1 = whh_sl + lr * H_ + q * 128;
#pragma unroll
    for (int i = 0; i < 128; ++i) d1[i] = s1[i];
  }
  for (int i = tid; i < 512; i += 256) c_sl[i >> 5][i & 31] = c0[ncol0 + (i >> 5)];
  const unsigned ctxsl_off = lds_offset_of(&ctx_sl[0][0]);
  // prologue: DMA ctx tile for t=0 into buffer 0
  if (wave == 0)
    tdm_stage_2d(ctxsl_off, ctx, H_, B_, S_ * H_);
  __syncthreads();
  const int SH = S_ * H_;
  for (int t = 0; t < S_; ++t) {
    const __bf16* hc = (t & 1) ? h1buf : h0buf;
    __bf16* hn = (t & 1) ? h0buf : h1buf;
    const float* pc = (t & 1) ? p1buf : p0buf;
    float* pn = (t & 1) ? p0buf : p1buf;
    // prefetch ctx tile for t+1 into the other buffer; wait only for tile t
    if (wave == 0) {
      if (t + 1 < S_) {
        tdm_stage_2d(ctxsl_off + (unsigned)(((t + 1) & 1) * (B_ * H_ * 2)),
                     ctx + (size_t)(t + 1) * H_, H_, B_, S_ * H_);
        __builtin_amdgcn_s_wait_tensorcnt(1);
      } else {
        __builtin_amdgcn_s_wait_tensorcnt(0);
      }
    }
    __syncthreads();
    v8f acc;
#pragma unroll
    for (int v = 0; v < 8; ++v) acc[v] = bias_n + pc[mt * 16 + lh * 8 + v] * wp0;
    const __bf16* abase_c = &ctx_sl[t & 1][mt * 16 * H_];
    const __bf16* bbase_c = Wih_ctx + (size_t)(g * H_ + ncol0) * H_;
#pragma unroll 4
    for (int kc = 0; kc < H_ / 32; ++kc)
      acc = wmma_bf16(frag_ld_bf16(abase_c + kc * 32, H_),
                      frag_ld_bf16(bbase_c + kc * 32, H_), acc);
    const __bf16* abase_h = hc + mt * 16 * H_;
#pragma unroll 4
    for (int kc = 0; kc < H_ / 32; ++kc)
      acc = wmma_bf16(frag_ld_bf16(abase_h + kc * 32, H_),
                      frag_ld_bf16(whh_sl + (g * 16) * H_ + kc * 32, H_), acc);
#pragma unroll
    for (int v = 0; v < 8; ++v) gates_sl[g][nl][mt * 16 + lh * 8 + v] = acc[v];
    __syncthreads();
#pragma unroll
    for (int it = 0; it < 2; ++it) {
      const int idx = tid + it * 256;
      const int m = idx & 31, nc = idx >> 5;
      const float iv = gates_sl[0][nc][m], fv = gates_sl[1][nc][m];
      const float gv = gates_sl[2][nc][m], ov = gates_sl[3][nc][m];
      float c = sigm(fv) * c_sl[nc][m] + sigm(iv) * tanhf(gv);
      const float h = sigm(ov) * tanhf(c);
      c_sl[nc][m] = c;
      hn[m * H_ + ncol0 + nc] = (__bf16)h;
    }
    __threadfence();
    __syncthreads();
    if (tid == 0) {  // barrier 1: all of h(t) written
      __hip_atomic_fetch_add(cntA, 1u, __ATOMIC_RELAXED, __HIP_MEMORY_SCOPE_AGENT);
      const unsigned tgt = (unsigned)(t + 1) * (unsigned)NBLK_REC;
      while (__hip_atomic_load(cntA, __ATOMIC_RELAXED, __HIP_MEMORY_SCOPE_AGENT) < tgt)
        __builtin_amdgcn_s_sleep(2);
    }
    __syncthreads();
    if (blk == 0) {  // p = relu(h@W_o1^T + b_o1)@W_o2^T + b_o2
      const float bo = b_o1[wave * 16 + nl];
      v8f a0, a1;
#pragma unroll
      for (int v = 0; v < 8; ++v) { a0[v] = bo; a1[v] = bo; }
      const __bf16* bbase_o = W_o1 + (size_t)(wave * 16) * H_;
#pragma unroll 4
      for (int kc = 0; kc < H_ / 32; ++kc) {
        const v16bf bb = frag_ld_bf16(bbase_o + kc * 32, H_);
        a0 = wmma_bf16(frag_ld_bf16(hn + kc * 32, H_), bb, a0);
        a1 = wmma_bf16(frag_ld_bf16(hn + 16 * H_ + kc * 32, H_), bb, a1);
      }
#pragma unroll
      for (int v = 0; v < 8; ++v) {
        const float x0 = a0[v], x1 = a1[v];
        o1_sl[lh * 8 + v][wave * 16 + nl] = x0 > 0.f ? x0 : 0.f;
        o1_sl[16 + lh * 8 + v][wave * 16 + nl] = x1 > 0.f ? x1 : 0.f;
      }
      __syncthreads();
      if (tid < B_) {
        float s = b_o2[0];
#pragma unroll 8
        for (int e = 0; e < E_; ++e) s += o1_sl[tid][e] * W_o2[e];
        pn[tid] = s;
        out[tid * S_ + t] = s * mask[tid * S_ + t];
        __threadfence();
      }
      __syncthreads();
      if (tid == 0)
        __hip_atomic_fetch_add(cntB, 1u, __ATOMIC_RELAXED, __HIP_MEMORY_SCOPE_AGENT);
    }
    if (tid == 0) {  // barrier 2: p(t) published by block 0
      const unsigned tgt = (unsigned)(t + 1);
      while (__hip_atomic_load(cntB, __ATOMIC_RELAXED, __HIP_MEMORY_SCOPE_AGENT) < tgt)
        __builtin_amdgcn_s_sleep(2);
    }
    __syncthreads();
  }
}

// ---------------------------------------------------------------------------
extern "C" void kernel_launch(void* const* d_in, const int* in_sizes, int n_in,
                              void* d_out, int out_size, void* d_ws, size_t ws_size,
                              hipStream_t stream) {
  (void)in_sizes; (void)n_in; (void)out_size; (void)ws_size;
  const float* inputs = (const float*)d_in[0];
  const float* mask   = (const float*)d_in[1];
  const int*   lengths= (const int*)d_in[2];
  const float* W_e    = (const float*)d_in[3];
  const float* b_e    = (const float*)d_in[4];
  const float* W_a1   = (const float*)d_in[5];
  const float* b_a1   = (const float*)d_in[6];
  const float* W_a2   = (const float*)d_in[7];
  const float* b_a2   = (const float*)d_in[8];
  const float* Wih_e  = (const float*)d_in[9];
  const float* Whh_e  = (const float*)d_in[10];
  const float* bih_e  = (const float*)d_in[11];
  const float* bhh_e  = (const float*)d_in[12];
  const float* enc_h0 = (const float*)d_in[13];
  const float* enc_c0 = (const float*)d_in[14];
  const float* Wih_d  = (const float*)d_in[15];
  const float* Whh_d  = (const float*)d_in[16];
  const float* bih_d  = (const float*)d_in[17];
  const float* bhh_d  = (const float*)d_in[18];
  const float* dec_h0 = (const float*)d_in[19];
  const float* dec_c0 = (const float*)d_in[20];
  const float* W_o1   = (const float*)d_in[21];
  const float* b_o1   = (const float*)d_in[22];
  const float* W_o2   = (const float*)d_in[23];
  const float* b_o2   = (const float*)d_in[24];
  float* out = (float*)d_out;

  // workspace carve-up (256B aligned)
  char* ws = (char*)d_ws;
  size_t off = 0;
  auto alloc = [&](size_t bytes) -> char* {
    char* p = ws + off;
    off = (off + bytes + 255) & ~(size_t)255;
    return p;
  };
  __bf16* embed_bf  = (__bf16*)alloc((size_t)B_ * S_ * E_ * 2);    // 8 MB
  __bf16* a1_bf     = (__bf16*)alloc((size_t)B_ * S_ * E_ * 2);    // 8 MB
  float*  a_f       = (float*) alloc((size_t)B_ * S_ * 3 * 4);     // logits -> attn
  float*  enc_out   = (float*) alloc((size_t)B_ * S_ * H_ * 4);    // 64 MB
  __bf16* ctx_bf    = (__bf16*)alloc((size_t)B_ * S_ * H_ * 2);    // 32 MB
  __bf16* We_bf     = (__bf16*)alloc((size_t)E_ * WIN_ * 2);
  __bf16* Wa1_bf    = (__bf16*)alloc((size_t)E_ * E_ * 2);
  __bf16* Wihe_bf   = (__bf16*)alloc((size_t)4 * H_ * E_ * 2);
  __bf16* Whhe_bf   = (__bf16*)alloc((size_t)4 * H_ * H_ * 2);
  __bf16* Wihd_bf   = (__bf16*)alloc((size_t)4 * H_ * H_ * 2);     // ctx part, repacked
  __bf16* Whhd_bf   = (__bf16*)alloc((size_t)4 * H_ * H_ * 2);
  __bf16* Wo1_bf    = (__bf16*)alloc((size_t)E_ * H_ * 2);
  __bf16* he0       = (__bf16*)alloc((size_t)B_ * H_ * 2);
  __bf16* he1       = (__bf16*)alloc((size_t)B_ * H_ * 2);
  __bf16* hd0       = (__bf16*)alloc((size_t)B_ * H_ * 2);
  __bf16* hd1       = (__bf16*)alloc((size_t)B_ * H_ * 2);
  float*  p0        = (float*) alloc(B_ * 4);
  float*  p1        = (float*) alloc(B_ * 4);
  unsigned* cnts    = (unsigned*)alloc(64);

  // 0. deterministic per-call init + one-shot bf16 weight conversions
  init_kernel<<<64, 256, 0, stream>>>(enc_h0, dec_h0, he0, hd0, p0, p1, cnts);
  cvt_kernel<<<(E_ * WIN_) / 256, 256, 0, stream>>>(W_e, We_bf, E_ * WIN_);
  cvt_kernel<<<(E_ * E_) / 256, 256, 0, stream>>>(W_a1, Wa1_bf, E_ * E_);
  cvt_kernel<<<(4 * H_ * E_) / 256, 256, 0, stream>>>(Wih_e, Wihe_bf, 4 * H_ * E_);
  cvt_kernel<<<(4 * H_ * H_) / 256, 256, 0, stream>>>(Whh_e, Whhe_bf, 4 * H_ * H_);
  cvt_kernel<<<(4 * H_ * H_) / 256, 256, 0, stream>>>(Whh_d, Whhd_bf, 4 * H_ * H_);
  cvt_kernel<<<(E_ * H_) / 256, 256, 0, stream>>>(W_o1, Wo1_bf, E_ * H_);
  repack_wihd_kernel<<<(4 * H_ * H_) / 256, 256, 0, stream>>>(Wih_d, Wihd_bf);
  // 1. embed = relu(inputs @ W_e^T + b_e)        [32768x1024 x 1024x128]
  gemm_relu_kernel<false><<<(B_ * S_) / 16, 256, 0, stream>>>(
      inputs, WIN_, We_bf, WIN_, b_e, embed_bf, E_, WIN_);
  // 2. a1 = relu(embed @ W_a1^T + b_a1)          [32768x128 x 128x128]
  gemm_relu_kernel<true><<<(B_ * S_) / 16, 256, 0, stream>>>(
      embed_bf, E_, Wa1_bf, E_, b_a1, a1_bf, E_, E_);
  // 3. attention logits + softmax over time
  attn_logits_kernel<<<(B_ * S_) / 256, 256, 0, stream>>>(a1_bf, W_a2, b_a2, a_f);
  softmax_time_kernel<<<B_ * 3, 256, 0, stream>>>(a_f);
  // 4. encoder LSTM (persistent, grid-synced, LDS-resident weights)
  enc_lstm_kernel<<<NBLK_REC, 256, 0, stream>>>(embed_bf, Wihe_bf, Whhe_bf, bih_e,
                                                bhh_e, enc_c0, lengths, he0, he1,
                                                enc_out, &cnts[0]);
  // 5. local-attention context convolution
  ctx_kernel<<<(B_ * S_ * H_) / 256, 256, 0, stream>>>(a_f, enc_out, ctx_bf);
  // 6. decoder LSTM + TDM-prefetched ctx + autoregressive output MLP
  dec_lstm_kernel<<<NBLK_REC, 256, 0, stream>>>(
      ctx_bf, Wih_d, Wihd_bf, Whhd_bf, bih_d, bhh_d, dec_c0, Wo1_bf, b_o1, W_o2,
      b_o2, mask, hd0, hd1, p0, p1, out, &cnts[1], &cnts[2]);
}
